// ChunkGatedDeltaRule_7945689498103
// MI455X (gfx1250) — matrix operand order repaired and checked by reading, not compiled, not run on hardware
//
#include <hip/hip_runtime.h>

// ---------------- problem constants (B,T,H,K,V fixed by the reference) ------
#define B_   4
#define T_   2048
#define H_   16
#define DK   128          // key dim
#define DV   128          // full value dim
#define DVL  64           // value slice per workgroup (V split 2-way)
#define VSPL (DV / DVL)
#define CH   64           // chunk length
#define NCH  (T_ / CH)
#define EPS_ 1e-6f

typedef __bf16 bf16_t;
typedef __attribute__((ext_vector_type(16))) __bf16 bf16x16;
typedef __attribute__((ext_vector_type(8)))  __bf16 bf16x8;
typedef __attribute__((ext_vector_type(8)))  float  f32x8;

#define F32X8_ZERO {0.f,0.f,0.f,0.f,0.f,0.f,0.f,0.f}

static __device__ __forceinline__ f32x8 wmma_bf16(f32x8 c, bf16x16 a, bf16x16 b) {
  // D = A(16x32 bf16) * B(32x16 bf16) + C(f32), RNE, f32 accumulate
  return __builtin_amdgcn_wmma_f32_16x16x32_bf16(false, a, false, b, (short)0, c,
                                                 false, false);
}

// A fragment, 16x32 tile of bf16 row-major [ld] matrix at (r0,k0).
// Lane L holds row r0+(L&15); K elements h*16..h*16+15, h=L>>4.
static __device__ __forceinline__ bf16x16 frag_a(const bf16_t* src, int ld,
                                                 int r0, int k0, int lane) {
  const int row  = r0 + (lane & 15);
  const int half = lane >> 4;
  const bf16_t* p = src + row * ld + k0 + half * 16;
  const bf16x8 lo = *(const bf16x8*)p;
  const bf16x8 hi = *(const bf16x8*)(p + 8);
  return __builtin_shufflevector(lo, hi, 0, 1, 2, 3, 4, 5, 6, 7,
                                         8, 9, 10, 11, 12, 13, 14, 15);
}

// B fragment when memory holds B^T row-major (mem[n][k], contiguous in k).
// Lane L holds column n0+(L&15); K = k0+(L>>4)*16 .. +15.
static __device__ __forceinline__ bf16x16 frag_bT(const bf16_t* src, int ld,
                                                  int k0, int n0, int lane) {
  const int n  = n0 + (lane & 15);
  const int kb = k0 + (lane >> 4) * 16;
  return *(const bf16x16*)(src + n * ld + kb);
}

// C fragment from an f32 matrix stored TRANSPOSED row-major (mem[n][m]).
static __device__ __forceinline__ f32x8 frag_cT_load(const float* src, int ld,
                                                     int r0, int c0, int lane) {
  const int n  = c0 + (lane & 15);
  const int rb = r0 + (lane >> 4) * 8;
  return *(const f32x8*)(src + n * ld + rb);
}

static __device__ __forceinline__ void store_bf16x8(bf16_t* p, f32x8 a) {
  bf16x8 v;
#pragma unroll
  for (int j = 0; j < 8; ++j) v[j] = (bf16_t)a[j];
  *(bf16x8*)p = v;
}

__global__ __launch_bounds__(256)
void gdn_chunk_wmma(const float* __restrict__ q, const float* __restrict__ k,
                    const float* __restrict__ v, const float* __restrict__ g,
                    const float* __restrict__ beta, const float* __restrict__ h0,
                    float* __restrict__ out) {
  extern __shared__ __attribute__((aligned(32))) char smem[];
  // f32 region
  float*  STf   = (float*)smem;        // [DVL][DK] state slice, TRANSPOSED f32
  float*  TfT   = STf + DVL * DK;      // [64][64]  (I+A)^-1, TRANSPOSED f32
  float*  gcum  = TfT + CH * CH;       // [64] within-chunk cumsum of g
  float*  bfac  = gcum + CH;           // [64] b_t = exp(gcum_t)
  float*  binv  = bfac + CH;           // [64] 1 / b_t
  float*  betas = binv + CH;           // [64]
  float*  ratio = betas + CH;          // [64] b_C / b_t
  float*  nbb   = ratio + CH;          // [64] -beta_t * b_t
  float*  rred  = nbb + CH;            // [512] l2norm partial sums (q|k)
  // bf16 region
  bf16_t* qb  = (bf16_t*)(rred + 512); // [64][128] normalized*scale q
  bf16_t* kb  = qb + CH * DK;          // [64][128] normalized k
  bf16_t* SbT = kb + CH * DK;          // [DVL][DK] bf16 mirror of state (T)
  bf16_t* kdT = SbT + DVL * DK;        // [128][64] (diag(ratio) K)^T row-major
  bf16_t* rbT = kdT + DK * CH;         // [DVL][64] rhs, TRANSPOSED
  bf16_t* ubT = rbT + DVL * CH;        // [DVL][64] U,   TRANSPOSED
  bf16_t* P0  = ubT + DVL * CH;        // [64][64] Neumann power, row-major
  bf16_t* P1  = P0 + CH * CH;
  bf16_t* PT0 = P1 + CH * CH;          // [64][64] Neumann power, transposed
  bf16_t* PT1 = PT0 + CH * CH;
  bf16_t* Tb  = PT1 + CH * CH;         // [64][64] bf16 T, row-major (A operand)
  bf16_t* Mb  = Tb + CH * CH;          // [64][64] M = tril(e^{dgc} q.k)

  const int bh   = blockIdx.x / VSPL;      // (b,h) pair
  const int vs   = blockIdx.x % VSPL;      // value slice
  const int b    = bh / H_, h = bh % H_;
  const int tid  = threadIdx.x;
  const int wave = tid >> 5, lane = tid & 31;
  const int vcol0 = vs * DVL;
  const float scale = 0.08838834764831845f;  // 128^-0.5
  const size_t rs = (size_t)H_ * DK;         // row stride in q/k/v/o (H*128)

  const float* qbase = q   + ((size_t)b * T_ * H_ + h) * DK;
  const float* kbase = k   + ((size_t)b * T_ * H_ + h) * DK;
  const float* vbase = v   + ((size_t)b * T_ * H_ + h) * DV;
  const float* gbase = g   + (size_t)b * T_ * H_ + h;
  const float* bbase = beta + (size_t)b * T_ * H_ + h;
  float*       obase = out + ((size_t)b * T_ * H_ + h) * DV;
  float*       fsb   = out + (size_t)B_ * T_ * H_ * DV + (size_t)bh * DK * DV;
  const float* h0p   = h0  + (size_t)bh * DK * DV;

  // ---- init state slice (transposed): STf[vv][kk] = h0[kk][vcol0+vv] ----
  for (int i = tid; i < DVL * DK; i += 256) {
    const int vv = i >> 7, kk = i & (DK - 1);
    const float s = h0p[(size_t)kk * DV + vcol0 + vv];
    STf[i] = s;
    SbT[i] = (bf16_t)s;
  }
  __syncthreads();

  for (int c = 0; c < NCH; ++c) {
    const int t0 = c * CH;

    // ---- P1: gates ----
    if (tid < CH) {
      gcum[tid]  = gbase[(size_t)(t0 + tid) * H_];
      betas[tid] = bbase[(size_t)(t0 + tid) * H_];
    }
    __syncthreads();
    if (tid == 0) {
      float a = 0.f;
      for (int i = 0; i < CH; ++i) { a += gcum[i]; gcum[i] = a; }
    }
    __syncthreads();
    if (tid < CH) {
      const float bt = expf(gcum[tid]);
      bfac[tid]  = bt;
      binv[tid]  = 1.0f / bt;
      ratio[tid] = expf(gcum[CH - 1] - gcum[tid]);
      nbb[tid]   = -betas[tid] * bt;
    }

    // ---- P2: load + l2norm q,k -> bf16 (4 threads per row) ----
    {
      const int row = tid >> 2, part = tid & 3;
      const float* qp = qbase + (size_t)(t0 + row) * rs + part * 32;
      const float* kp = kbase + (size_t)(t0 + row) * rs + part * 32;
      float qv[32], kv[32], sq = 0.f, sk = 0.f;
#pragma unroll
      for (int i = 0; i < 32; ++i) { qv[i] = qp[i]; sq += qv[i] * qv[i]; }
#pragma unroll
      for (int i = 0; i < 32; ++i) { kv[i] = kp[i]; sk += kv[i] * kv[i]; }
      rred[row * 4 + part]       = sq;
      rred[256 + row * 4 + part] = sk;
      if (c + 1 < NCH) {  // pull next chunk toward the caches
        __builtin_prefetch(qp + (size_t)CH * rs, 0, 0);
        __builtin_prefetch(kp + (size_t)CH * rs, 0, 0);
      }
      __syncthreads();
      const float qs = rred[row*4] + rred[row*4+1] + rred[row*4+2] + rred[row*4+3];
      const float ks = rred[256+row*4] + rred[256+row*4+1] + rred[256+row*4+2] + rred[256+row*4+3];
      const float qn = rsqrtf(qs + EPS_) * scale;
      const float kn = rsqrtf(ks + EPS_);
      bf16_t* qd = qb + row * DK + part * 32;
      bf16_t* kd = kb + row * DK + part * 32;
#pragma unroll
      for (int i = 0; i < 32; ++i) {
        qd[i] = (bf16_t)(qv[i] * qn);
        kd[i] = (bf16_t)(kv[i] * kn);
      }
    }
    __syncthreads();

    // ---- P3: kdT precompute + N = -tril_s(beta e^{dgc} k.k), M = tril(e^{dgc} q.k)
    // kdT[kk][s] = ratio[s] * k_s[kk] ; 8 consecutive s per thread
    for (int grp = tid; grp < DK * CH / 8; grp += 256) {
      const int kk = grp >> 3, s0 = (grp & 7) * 8;
      const f32x8 r8 = *(const f32x8*)(ratio + s0);
      f32x8 a;
#pragma unroll
      for (int j = 0; j < 8; ++j) a[j] = (float)kb[(s0 + j) * DK + kk] * r8[j];
      store_bf16x8(kdT + kk * CH + s0, a);
    }
    for (int tile = wave; tile < 32; tile += 8) {
      const int mtx = tile >> 4;                 // 0 -> N (K.K), 1 -> M (Q.K)
      const int tr = (tile >> 2) & 3, tc = tile & 3;
      const int r0 = tr * 16, c0 = tc * 16;
      f32x8 acc = F32X8_ZERO;
      if (tr >= tc) {
        const bf16_t* asrc = mtx ? qb : kb;
#pragma unroll
        for (int kk = 0; kk < DK; kk += 32)
          acc = wmma_bf16(acc, frag_a(asrc, DK, r0, kk, lane),
                               frag_bT(kb, DK, kk, c0, lane));
      }
      const int n = c0 + (lane & 15), rb8 = r0 + (lane >> 4) * 8;
      const float bn = binv[n];  // e^{gc_r - gc_n} = bfac[r] * binv[n]
      if (mtx) {
        const f32x8 bf8 = *(const f32x8*)(bfac + rb8);  // in regs -> cndmask
#pragma unroll
        for (int j = 0; j < 8; ++j) {
          const float val = (n <= rb8 + j) ? bf8[j] * bn * acc[j] : 0.f;
          Mb[(rb8 + j) * CH + n] = (bf16_t)val;
        }
      } else {
        const f32x8 nb8 = *(const f32x8*)(nbb + rb8);
        f32x8 col;
#pragma unroll
        for (int j = 0; j < 8; ++j) {
          const float val = (n < rb8 + j) ? nb8[j] * bn * acc[j] : 0.f;
          P0[(rb8 + j) * CH + n] = (bf16_t)val;   // N row-major (A operand)
          col[j] = val;
        }
        store_bf16x8(PT0 + n * CH + rb8, col);    // N transposed (B operand)
      }
    }
    __syncthreads();

    // ---- P4: T = (I - N)^{-1} = prod(I + N^{2^i}) (N nilpotent, N^64 = 0) ----
    for (int i = tid; i < CH * CH; i += 256) {
      const int rr = i >> 6, cc = i & 63;
      const float val = (rr == cc ? 1.f : 0.f) + (float)P0[i];
      TfT[cc * CH + rr] = val;       // transposed f32 (C operand)
      Tb[i] = (bf16_t)val;           // row-major bf16 (A operand)
    }
    __syncthreads();
    bf16_t* Pc = P0;  bf16_t* PTc = PT0;
    bf16_t* Pn = P1;  bf16_t* PTn = PT1;
    for (int it = 0; it < 5; ++it) {
      // square the power: Pn = Pc @ Pc (strictly lower result)
      for (int tile = wave; tile < 16; tile += 8) {
        const int tr = tile >> 2, tc = tile & 3;
        const int r0 = tr * 16, c0 = tc * 16;
        f32x8 acc = F32X8_ZERO;
        if (tr >= tc) {
#pragma unroll
          for (int kk = 0; kk < CH; kk += 32)
            acc = wmma_bf16(acc, frag_a(Pc, CH, r0, kk, lane),
                                 frag_bT(PTc, CH, kk, c0, lane));
        }
        const int n = c0 + (lane & 15), rb8 = r0 + (lane >> 4) * 8;
#pragma unroll
        for (int j = 0; j < 8; ++j) Pn[(rb8 + j) * CH + n] = (bf16_t)acc[j];
        store_bf16x8(PTn + n * CH + rb8, acc);
      }
      __syncthreads();
      // T <- T + T @ Pn  (two-phase: compute to regs, barrier, store)
      f32x8 upd[2];
#pragma unroll
      for (int i2 = 0; i2 < 2; ++i2) {
        const int tile = wave + i2 * 8;
        const int tr = tile >> 2, tc = tile & 3;
        const int r0 = tr * 16, c0 = tc * 16;
        f32x8 acc = frag_cT_load(TfT, CH, r0, c0, lane);
        if (tr >= tc) {
#pragma unroll
          for (int kk = 0; kk < CH; kk += 32)
            acc = wmma_bf16(acc, frag_a(Tb, CH, r0, kk, lane),
                                 frag_bT(PTn, CH, kk, c0, lane));
        }
        upd[i2] = acc;
      }
      __syncthreads();
#pragma unroll
      for (int i2 = 0; i2 < 2; ++i2) {
        const int tile = wave + i2 * 8;
        const int tr = tile >> 2, tc = tile & 3;
        const int n = tc * 16 + (lane & 15), rb8 = tr * 16 + (lane >> 4) * 8;
        *(f32x8*)(TfT + n * CH + rb8) = upd[i2];
#pragma unroll
        for (int j = 0; j < 8; ++j) Tb[(rb8 + j) * CH + n] = (bf16_t)upd[i2][j];
      }
      __syncthreads();
      bf16_t* t1 = Pc;  Pc  = Pn;  Pn  = t1;
      bf16_t* t2 = PTc; PTc = PTn; PTn = t2;
    }

    // ---- P5: rhs = beta .* v + nbb .* (K @ S0)  (stored transposed) ----
    for (int tile = wave; tile < 16; tile += 8) {
      const int tr = tile >> 2, tc = tile & 3;
      const int r0 = tr * 16, c0 = tc * 16;
      f32x8 acc = F32X8_ZERO;
#pragma unroll
      for (int kk = 0; kk < DK; kk += 32)
        acc = wmma_bf16(acc, frag_a(kb, DK, r0, kk, lane),
                             frag_bT(SbT, DK, kk, c0, lane));
      const int n = c0 + (lane & 15), rb8 = r0 + (lane >> 4) * 8;
      const f32x8 be8 = *(const f32x8*)(betas + rb8);
      const f32x8 nb8 = *(const f32x8*)(nbb + rb8);
      f32x8 col;
#pragma unroll
      for (int j = 0; j < 8; ++j) {
        const float vv = vbase[(size_t)(t0 + rb8 + j) * rs + vcol0 + n];
        col[j] = be8[j] * vv + nb8[j] * acc[j];
      }
      store_bf16x8(rbT + n * CH + rb8, col);
    }
    __syncthreads();

    // ---- P6: U = T @ rhs  (stored transposed) ----
    for (int tile = wave; tile < 16; tile += 8) {
      const int tr = tile >> 2, tc = tile & 3;
      const int r0 = tr * 16, c0 = tc * 16;
      f32x8 acc = F32X8_ZERO;
#pragma unroll
      for (int kk = 0; kk < CH; kk += 32)
        acc = wmma_bf16(acc, frag_a(Tb, CH, r0, kk, lane),
                             frag_bT(rbT, CH, kk, c0, lane));
      const int n = c0 + (lane & 15), rb8 = r0 + (lane >> 4) * 8;
      store_bf16x8(ubT + n * CH + rb8, acc);
    }
    __syncthreads();

    // ---- P7: O = b .* (Q @ S0) + M @ U  -> global ----
    for (int tile = wave; tile < 16; tile += 8) {
      const int tr = tile >> 2, tc = tile & 3;
      const int r0 = tr * 16, c0 = tc * 16;
      f32x8 acc = F32X8_ZERO;
#pragma unroll
      for (int kk = 0; kk < DK; kk += 32)
        acc = wmma_bf16(acc, frag_a(qb, DK, r0, kk, lane),
                             frag_bT(SbT, DK, kk, c0, lane));
      const int n = c0 + (lane & 15), rb8 = r0 + (lane >> 4) * 8;
      const f32x8 bf8 = *(const f32x8*)(bfac + rb8);
#pragma unroll
      for (int j = 0; j < 8; ++j) acc[j] *= bf8[j];
#pragma unroll
      for (int kk = 0; kk < CH; kk += 32)
        acc = wmma_bf16(acc, frag_a(Mb, CH, r0, kk, lane),
                             frag_bT(ubT, CH, kk, c0, lane));
#pragma unroll
      for (int j = 0; j < 8; ++j)
        obase[(size_t)(t0 + rb8 + j) * rs + vcol0 + n] = acc[j];
    }
    __syncthreads();

    // ---- P8: S = b_C * S + (diag(b_C/b_s) K)^T @ U  (tile-exclusive, in place)
    const float bC = bfac[CH - 1];
    for (int tile = wave; tile < 32; tile += 8) {
      const int tr = tile >> 2, tc = tile & 3;       // 8 x 4 grid over [128][64]
      const int r0 = tr * 16, c0 = tc * 16;
      f32x8 acc = frag_cT_load(STf, DK, r0, c0, lane);
#pragma unroll
      for (int j = 0; j < 8; ++j) acc[j] *= bC;
#pragma unroll
      for (int kk = 0; kk < CH; kk += 32)
        acc = wmma_bf16(acc, frag_a(kdT, CH, r0, kk, lane),
                             frag_bT(ubT, CH, kk, c0, lane));
      const int n = c0 + (lane & 15), rb8 = r0 + (lane >> 4) * 8;
      *(f32x8*)(STf + n * DK + rb8) = acc;
      store_bf16x8(SbT + n * DK + rb8, acc);
    }
    __syncthreads();
  }

  // ---- final state out (undo transpose) ----
  for (int i = tid; i < DVL * DK; i += 256) {
    const int vv = i >> 7, kk = i & (DK - 1);
    fsb[(size_t)kk * DV + vcol0 + vv] = STf[i];
  }
}

extern "C" void kernel_launch(void* const* d_in, const int* in_sizes, int n_in,
                              void* d_out, int out_size, void* d_ws, size_t ws_size,
                              hipStream_t stream) {
  (void)in_sizes; (void)n_in; (void)out_size; (void)d_ws; (void)ws_size;
  const float* q    = (const float*)d_in[0];
  const float* k    = (const float*)d_in[1];
  const float* v    = (const float*)d_in[2];
  const float* g    = (const float*)d_in[3];
  const float* beta = (const float*)d_in[4];
  const float* h0   = (const float*)d_in[5];
  float* out = (float*)d_out;

  constexpr size_t kFloats = (size_t)DVL * DK + CH * CH + 6 * CH + 512;
  constexpr size_t kBf16s  = (size_t)2 * CH * DK      // qb, kb
                           + (size_t)DVL * DK         // SbT
                           + (size_t)DK * CH          // kdT
                           + (size_t)2 * DVL * CH     // rbT, ubT
                           + (size_t)6 * CH * CH;     // P0,P1,PT0,PT1,Tb,Mb
  constexpr size_t kSmem   = kFloats * sizeof(float) + kBf16s * sizeof(bf16_t);

  (void)hipFuncSetAttribute(reinterpret_cast<const void*>(gdn_chunk_wmma),
                            hipFuncAttributeMaxDynamicSharedMemorySize,
                            (int)kSmem);
  gdn_chunk_wmma<<<B_ * H_ * VSPL, 256, kSmem, stream>>>(q, k, v, g, beta, h0, out);
}
